// HyperNetwork_5918464934269
// MI455X (gfx1250) — matrix operand order repaired and checked by reading, not compiled, not run on hardware
//
#include <hip/hip_runtime.h>
#include <hip/hip_bf16.h>
#include <math.h>

// ---------------- problem constants (match reference) ----------------
constexpr int Bsz = 64;      // batch
constexpr int IN  = 256;     // IN_DIM
constexpr int Hd  = 3072;    // hidden
constexpr int Od  = 1536;    // MLP output dim
constexpr int Ll  = 2;       // hidden layers
constexpr int KC  = 4;       // split-K chunks

// ---------------- WMMA types ----------------
typedef __bf16 v16bf __attribute__((ext_vector_type(16)));
typedef float  v8f   __attribute__((ext_vector_type(8)));

union FragBF { v16bf v; unsigned int u[8]; };

// Four (target, r) streams share one launch via blockIdx.z.
struct GemmArgs { const __bf16* X[4]; const float* W[4]; };
struct RedArgs  { const float* Bs[4]; void* Y[4]; };

// ---------------- CDNA5 async global->LDS (ASYNCcnt path) ----------------
__device__ __forceinline__ void async_lds_b128(unsigned lds_off, const __bf16* g)
{
    unsigned long long ga = (unsigned long long)(const void*)g;
    asm volatile("global_load_async_to_lds_b128 %0, %1, off"
                 :: "v"(lds_off), "v"(ga) : "memory");
}
__device__ __forceinline__ void wait_async0()
{
    asm volatile("s_wait_asynccnt 0x0" ::: "memory");
}

// ---------------- tiny f32 -> bf16 conversion ----------------
__global__ __launch_bounds__(256) void f32_to_bf16(const float* __restrict__ s,
                                                   __bf16* __restrict__ d, int n)
{
    int i = blockIdx.x * 256 + threadIdx.x;
    if (i < n) d[i] = (__bf16)s[i];
}

// ---------------- split-K partial GEMM: P = X[64,Kc] @ W[:,Kc]^T ----------------
// Block: 256 threads = 8 waves. Tile M=64 x N=128, K-step 32 (bf16 WMMA).
// Double-buffered LDS; W tiles register-pipelined (f32->bf16 cvt), A tiles
// copied with global_load_async_to_lds_b128.
__global__ __launch_bounds__(256) void gemm_partial(GemmArgs args, float* __restrict__ P,
                                                    int K, int Kc, int N)
{
    constexpr int KT = 32;
    constexpr int NB = 128;

    const int s = blockIdx.z;                 // stream (target*2 + r)
    const int p = blockIdx.y;                 // k-chunk
    const __bf16* __restrict__ X = args.X[s];
    const float*  __restrict__ W = args.W[s];

    __shared__ __attribute__((aligned(16))) __bf16 lA[2][64 * KT];
    __shared__ __attribute__((aligned(16))) __bf16 lB[2][NB * KT];

    const int tid  = threadIdx.x;
    const int lane = tid & 31;
    const int wave = tid >> 5;
    const int rlow = lane & 15;
    const int hi   = lane >> 4;
    const int n0   = blockIdx.x * NB;
    const int kbeg = p * Kc;

    const float*  wrow = W + (size_t)(n0 + (tid >> 1)) * K + (tid & 1) * 16 + kbeg;
    const __bf16* xrow = X + (size_t)(tid >> 2) * K + (tid & 3) * 8 + kbeg;

    float4 w0, w1, w2, w3;                    // W staging registers

    auto loadW = [&](int k0) {
        const float4* s4 = (const float4*)(wrow + k0);
        w0 = s4[0]; w1 = s4[1]; w2 = s4[2]; w3 = s4[3];
        if (k0 + 2 * KT < Kc)
            __builtin_prefetch(wrow + k0 + 2 * KT, 0, 1);  // global_prefetch_b8
    };
    auto storeW = [&](int b) {
        __bf16* dst = &lB[b][(tid >> 1) * KT + (tid & 1) * 16];
        const float f[16] = { w0.x,w0.y,w0.z,w0.w, w1.x,w1.y,w1.z,w1.w,
                              w2.x,w2.y,w2.z,w2.w, w3.x,w3.y,w3.z,w3.w };
        #pragma unroll
        for (int e = 0; e < 16; ++e) dst[e] = (__bf16)f[e];
    };
    auto issueA = [&](int k0, int b) {
        unsigned off = (unsigned)(unsigned long long)(const void*)
            &lA[b][(tid >> 2) * KT + (tid & 3) * 8];
        async_lds_b128(off, xrow + k0);       // 16B/lane raw bf16 copy
    };

    v8f acc[4] = {};
    const int nk = Kc / KT;

    loadW(0);
    issueA(0, 0);
    storeW(0);
    wait_async0();

    int buf = 0;
    for (int kt = 0; kt < nk; ++kt) {
        __syncthreads();                      // buf fully staged for all waves
        const bool more = (kt + 1) < nk;
        if (more) { loadW((kt + 1) * KT); issueA((kt + 1) * KT, buf ^ 1); }

        // B fragment (32x16 bf16): col = lane&15, K half picked by lane group
        FragBF bfr;
        {
            const unsigned int* lBu = (const unsigned int*)lB[buf];
            const int nloc = wave * 16 + rlow;
            #pragma unroll
            for (int v = 0; v < 8; ++v)
                bfr.u[v] = lBu[nloc * (KT / 2) + hi * 8 + v];
        }
        // A fragments + 4 WMMAs (B reused)
        const unsigned int* lAu = (const unsigned int*)lA[buf];
        #pragma unroll
        for (int m = 0; m < 4; ++m) {
            FragBF af;
            const int arow = m * 16 + rlow;
            #pragma unroll
            for (int v = 0; v < 8; ++v) {
                const int kst = 2 * v + hi * 16;
                af.u[v] = lAu[arow * (KT / 2) + (kst >> 1)];
            }
            acc[m] = __builtin_amdgcn_wmma_f32_16x16x32_bf16(
                false, af.v, false, bfr.v, (short)0, acc[m], false, false);
        }

        if (more) { storeW(buf ^ 1); wait_async0(); }  // overlaps HBM latency w/ WMMA
        buf ^= 1;
    }

    // partial f32 store (stays resident in 192MB L2)
    const int ng = n0 + wave * 16 + rlow;
    float* Pd = P + (size_t)(p * 4 + s) * 64 * N;
    #pragma unroll
    for (int m = 0; m < 4; ++m) {
        #pragma unroll
        for (int rr = 0; rr < 8; ++rr) {
            const int mrow = m * 16 + rr + hi * 8;    // ISA C/D layout
            Pd[(size_t)mrow * N + ng] = acc[m][rr];
        }
    }
}

// ---------------- reduce split-K partials + bias + exact GeLU ----------------
__global__ __launch_bounds__(256) void reduce_bias_act(const float* __restrict__ P,
                                                       RedArgs ra, int N, int act)
{
    const int s = blockIdx.y;
    const int cnt4 = (Bsz * N) >> 2;
    const int i = blockIdx.x * 256 + threadIdx.x;
    if (i >= cnt4) return;

    const size_t sN4 = ((size_t)Bsz * N) >> 2;
    const float4* P4 = (const float4*)P;
    float4 a = {0.f, 0.f, 0.f, 0.f};
    #pragma unroll
    for (int pp = 0; pp < KC; ++pp) {
        const float4 v = P4[(size_t)(pp * 4 + s) * sN4 + i];
        a.x += v.x; a.y += v.y; a.z += v.z; a.w += v.w;
    }
    const int n = (i << 2) % N;
    const float4 bv = *(const float4*)(ra.Bs[s] + n);
    float y[4] = { a.x + bv.x, a.y + bv.y, a.z + bv.z, a.w + bv.w };

    if (act) {
        __bf16* d = (__bf16*)ra.Y[s] + ((size_t)i << 2);
        #pragma unroll
        for (int c = 0; c < 4; ++c) {
            const float g = 0.5f * y[c] * (1.0f + erff(y[c] * 0.70710678118654752f));
            d[c] = (__bf16)g;
        }
    } else {
        ((float4*)ra.Y[s])[i] = make_float4(y[0], y[1], y[2], y[3]);
    }
}

// ---------------- epilogue: out[b,i,j] = base[i,j] + sum_r v1[r,b,i]*v2[r,b,j] ----
__global__ __launch_bounds__(256) void outer_update(const float* __restrict__ flat,
                                                    const float* __restrict__ base,
                                                    float* __restrict__ out,
                                                    int d1, int d2)
{
    const int b   = blockIdx.y;
    const int nj  = d2 >> 2;
    const int per = d1 * nj;
    const int idx = blockIdx.x * 256 + threadIdx.x;
    if (idx >= per) return;
    const int i  = idx / nj;
    const int jv = idx - i * nj;

    const float* f0 = flat + (size_t)b * Od;
    const float* f1 = flat + (size_t)Bsz * Od + (size_t)b * Od;
    const float s0 = f0[i];
    const float s1 = f1[i];
    const float4 v20 = ((const float4*)(f0 + d1))[jv];
    const float4 v21 = ((const float4*)(f1 + d1))[jv];
    const float4 bs  = ((const float4*)base)[(size_t)i * nj + jv];

    float4 o;
    o.x = fmaf(s1, v21.x, fmaf(s0, v20.x, bs.x));
    o.y = fmaf(s1, v21.y, fmaf(s0, v20.y, bs.y));
    o.z = fmaf(s1, v21.z, fmaf(s0, v20.z, bs.z));
    o.w = fmaf(s1, v21.w, fmaf(s0, v20.w, bs.w));
    ((float4*)out)[(size_t)b * per + idx] = o;
}

// ---------------- host-side orchestration ----------------
extern "C" void kernel_launch(void* const* d_in, const int* in_sizes, int n_in,
                              void* d_out, int out_size, void* d_ws, size_t ws_size,
                              hipStream_t stream)
{
    const float* u0 = (const float*)d_in[0];
    const float* basew[2] = { (const float*)d_in[1], (const float*)d_in[2] };
    const float *win[2], *bin[2], *hw[2], *hb[2], *wout[2], *bout[2];
    for (int t = 0; t < 2; ++t) {
        win [t] = (const float*)d_in[3 + t * 6 + 0];
        bin [t] = (const float*)d_in[3 + t * 6 + 1];
        hw  [t] = (const float*)d_in[3 + t * 6 + 2];
        hb  [t] = (const float*)d_in[3 + t * 6 + 3];
        wout[t] = (const float*)d_in[3 + t * 6 + 4];
        bout[t] = (const float*)d_in[3 + t * 6 + 5];
    }

    // ---- workspace layout (bytes) ----
    char* ws = (char*)d_ws;
    __bf16* u0b = (__bf16*)ws;                                   // 64*256 bf16 (32KB)
    __bf16* hA  = (__bf16*)(ws + 32768);                         // 4*64*3072 bf16
    __bf16* hB  = hA + (size_t)4 * Bsz * Hd;                     // 4*64*3072 bf16
    float*  flat = (float*)(ws + 3178496);                       // 4*64*1536 f32
    float*  P    = (float*)(ws + 4751360);                       // 16*64*3072 f32 partials

    const dim3 blk(256);
    GemmArgs ga;
    RedArgs  ra;

    // ---- u0 -> bf16 ----
    f32_to_bf16<<<dim3((Bsz * IN + 255) / 256), blk, 0, stream>>>(u0, u0b, Bsz * IN);

    // ---- L0: (64x256) @ win^T -> gelu -> hA (bf16) ----
    for (int s = 0; s < 4; ++s) {
        const int t = s >> 1, r = s & 1;
        ga.X[s] = u0b;
        ga.W[s] = win[t] + (size_t)r * Hd * IN;
        ra.Bs[s] = bin[t] + (size_t)r * Hd;
        ra.Y[s]  = hA + (size_t)s * Bsz * Hd;
    }
    gemm_partial<<<dim3(Hd / 128, KC, 4), blk, 0, stream>>>(ga, P, IN, IN / KC, Hd);
    reduce_bias_act<<<dim3((Bsz * Hd / 4 + 255) / 256, 4), blk, 0, stream>>>(P, ra, Hd, 1);

    // ---- L1, L2: (64x3072) @ hw[l]^T -> gelu ----
    for (int l = 0; l < Ll; ++l) {
        __bf16* src = (l == 0) ? hA : hB;
        __bf16* dst = (l == 0) ? hB : hA;
        for (int s = 0; s < 4; ++s) {
            const int t = s >> 1, r = s & 1;
            ga.X[s] = src + (size_t)s * Bsz * Hd;
            ga.W[s] = hw[t] + (size_t)(r * Ll + l) * Hd * Hd;
            ra.Bs[s] = hb[t] + (size_t)(r * Ll + l) * Hd;
            ra.Y[s]  = dst + (size_t)s * Bsz * Hd;
        }
        gemm_partial<<<dim3(Hd / 128, KC, 4), blk, 0, stream>>>(ga, P, Hd, Hd / KC, Hd);
        reduce_bias_act<<<dim3((Bsz * Hd / 4 + 255) / 256, 4), blk, 0, stream>>>(P, ra, Hd, 1);
    }

    // ---- L3: (64x3072) @ wout^T -> flat (f32, no act) ----
    for (int s = 0; s < 4; ++s) {
        const int t = s >> 1, r = s & 1;
        ga.X[s] = hA + (size_t)s * Bsz * Hd;
        ga.W[s] = wout[t] + (size_t)r * Od * Hd;
        ra.Bs[s] = bout[t] + (size_t)r * Od;
        ra.Y[s]  = flat + (size_t)s * Bsz * Od;
    }
    gemm_partial<<<dim3(Od / 128, KC, 4), blk, 0, stream>>>(ga, P, Hd, Hd / KC, Od);
    reduce_bias_act<<<dim3((Bsz * Od / 4 + 255) / 256, 4), blk, 0, stream>>>(P, ra, Od, 0);

    // ---- outer product update per target ----
    for (int t = 0; t < 2; ++t) {
        const int d1 = (t == 0) ? 1024 : 512;
        const int d2 = (t == 0) ? 512 : 1024;
        const float* flat_t = flat + (size_t)t * 2 * Bsz * Od;
        float* out_t = (float*)d_out + (size_t)t * Bsz * 1024 * 512;
        const int per4 = d1 * (d2 >> 2);
        outer_update<<<dim3((per4 + 255) / 256, Bsz), blk, 0, stream>>>(
            flat_t, basew[t], out_t, d1, d2);
    }
}
